// GreedyBatchedRNNTLoopLabelsComputer_29927332118545
// MI455X (gfx1250) — matrix operand order repaired and checked by reading, not compiled, not run on hardware
//
#include <hip/hip_runtime.h>

// ---------------- problem dims ----------------
#define BB    32
#define TT    64
#define ENC   1024
#define HD    320
#define JD    320
#define VV    1024          // BLANK = VV
#define NOUT  1025
#define NOUTP 1040          // padded to 65 N-tiles
#define SS    128           // scan steps

typedef __attribute__((ext_vector_type(16))) __bf16 v16bf;
typedef __attribute__((ext_vector_type(8)))  float  v8f;

__device__ __forceinline__ v8f wmma_bf16(v16bf a, v16bf b, v8f c) {
  return __builtin_amdgcn_wmma_f32_16x16x32_bf16(false, a, false, b, (short)0, c, false, false);
}

// A-matrix 16x32 bf16 fragment: lane L (m = L&15, half = L>>4) holds 16 values,
// element i maps to k = ((i>>3)<<4) | (half<<3) | (i&7)   (ISA 7.12.2)
__device__ __forceinline__ int kidxA(int half, int i) {
  return ((i >> 3) << 4) | (half << 3) | (i & 7);
}

__device__ __forceinline__ float sigm(float x) { return 1.f / (1.f + expf(-x)); }

// ---------------- prep kernels ----------------
// concat [W_ih; W_hh] -> (640 x 1280) f32 staging
__global__ void concat_lstm_kernel(const float* __restrict__ W_ih,
                                   const float* __restrict__ W_hh,
                                   float* __restrict__ dst) {
  long total = 640L * 1280L;
  for (long e = (long)blockIdx.x * blockDim.x + threadIdx.x; e < total;
       e += (long)gridDim.x * blockDim.x) {
    int r = (int)(e / 1280), c = (int)(e % 1280);
    dst[e] = (r < HD) ? W_ih[(long)r * 1280 + c] : W_hh[(long)(r - HD) * 1280 + c];
  }
}

// pack a (K x N) f32 matrix into B-fragment bf16 tiles, tile order nt*Kt + kt.
// lane L holds column n = nt*16 + (L&15), rows k = kt*32 + kidxA(L>>4, i).
__global__ void pack_b_kernel(const float* __restrict__ src, __bf16* __restrict__ dst,
                              int K, int N, int Npad) {
  int Kt = K >> 5, Nt = Npad >> 4;
  long total = (long)Kt * Nt * 512;
  for (long e = (long)blockIdx.x * blockDim.x + threadIdx.x; e < total;
       e += (long)gridDim.x * blockDim.x) {
    int i = (int)(e & 15); int lane = (int)((e >> 4) & 31); long tile = e >> 9;
    int kt = (int)(tile % Kt); int nt = (int)(tile / Kt);
    int k = kt * 32 + kidxA(lane >> 4, i);
    int n = nt * 16 + (lane & 15);
    float v = (n < N) ? src[(long)k * N + n] : 0.f;
    dst[e] = (__bf16)v;
  }
}

// pack a (M x K) f32 matrix into A-fragment bf16 tiles, tile order mt*Kt + kt.
__global__ void pack_a_kernel(const float* __restrict__ src, __bf16* __restrict__ dst,
                              int M, int K) {
  int Kt = K >> 5, Mt = M >> 4;
  long total = (long)Kt * Mt * 512;
  for (long e = (long)blockIdx.x * blockDim.x + threadIdx.x; e < total;
       e += (long)gridDim.x * blockDim.x) {
    int i = (int)(e & 15); int lane = (int)((e >> 4) & 31); long tile = e >> 9;
    int kt = (int)(tile % Kt); int mt = (int)(tile / Kt);
    int m = mt * 16 + (lane & 15);
    int k = kt * 32 + kidxA(lane >> 4, i);
    dst[e] = (__bf16)src[(long)m * K + k];
  }
}

// ---------------- encoder GEMM: x_proj = x @ W_enc + b_enc ----------------
// M=2048 (128 Mtiles), N=320 (20 Ntiles), K=1024 (32 Ktiles). 1 tile per wave.
__global__ void enc_gemm_kernel(const __bf16* __restrict__ packX,
                                const __bf16* __restrict__ packW,
                                const float* __restrict__ b_enc,
                                float* __restrict__ xproj) {
  const int Kt = 32, Nt = 20;
  int lane = threadIdx.x & 31, wave = threadIdx.x >> 5;
  int tile = blockIdx.x * 8 + wave;           // 2560 tiles total, 320 blocks x 8 waves
  if (tile >= 128 * Nt) return;               // wave-uniform exit
  int mt = tile / Nt, nt = tile % Nt;
  const v16bf* A = (const v16bf*)packX;
  const v16bf* Bm = (const v16bf*)packW;
  v8f acc = {};
  for (int kt = 0; kt < Kt; ++kt) {
    v16bf a = A[(mt * Kt + kt) * 32 + lane];
    v16bf b = Bm[(nt * Kt + kt) * 32 + lane];
    acc = wmma_bf16(a, b, acc);
  }
  int n = nt * 16 + (lane & 15);
  float bias = b_enc[n];
  int mbase = mt * 16 + ((lane >> 4) << 3);
  for (int r = 0; r < 8; ++r)
    xproj[(long)(mbase + r) * JD + n] = acc[r] + bias;
}

// ---------------- persistent decode kernel (1 block, 16 waves) ----------------
__device__ __forceinline__ void pred_update(const float* sH, float* sDec, v16bf* sPackA,
                                            const __bf16* pWpred, const float* b_pred,
                                            int tid, int lane, int wave) {
  __bf16* packS = (__bf16*)sPackA;
  // pack A = h (32 x 320), Kt = 10
  for (int e = tid; e < 2 * 10 * 512; e += 512) {
    int i = e & 15, ln = (e >> 4) & 31, tile = e >> 9;
    int kt = tile % 10, mt = tile / 10;
    int m = mt * 16 + (ln & 15);
    int k = kt * 32 + kidxA(ln >> 4, i);
    packS[e] = (__bf16)sH[m * HD + k];
  }
  __syncthreads();
  for (int t = wave; t < 2 * 20; t += 16) {          // wave-uniform tile loop
    int mt = t & 1, nt = t >> 1;
    v8f acc = {};
    for (int kt = 0; kt < 10; ++kt) {
      v16bf a = sPackA[(mt * 10 + kt) * 32 + lane];
      v16bf b = ((const v16bf*)pWpred)[(nt * 10 + kt) * 32 + lane];
      acc = wmma_bf16(a, b, acc);
    }
    int n = nt * 16 + (lane & 15);
    float bias = b_pred[n];
    int mbase = mt * 16 + ((lane >> 4) << 3);
    for (int r = 0; r < 8; ++r)
      sDec[(mbase + r) * JD + n] = acc[r] + bias;
  }
  __syncthreads();
}

__launch_bounds__(512, 1)
__global__ void decode_kernel(const float* __restrict__ xproj,
                              const int* __restrict__ out_len,
                              const float* __restrict__ Emb,
                              const float* __restrict__ b_lstm,
                              const float* __restrict__ b_pred,
                              const float* __restrict__ b_out,
                              const __bf16* __restrict__ pWout,
                              const __bf16* __restrict__ pWlstm,
                              const __bf16* __restrict__ pWpred,
                              int* __restrict__ lab_out, float* __restrict__ sc_out,
                              int* __restrict__ tm_out, int* __restrict__ len_out) {
  __shared__ float sDec[BB * JD];                // 40 KB
  __shared__ float sH[BB * HD];                  // 40 KB
  __shared__ float sC[BB * HD];                  // 40 KB
  __shared__ __bf16 sGates[BB * 4 * HD];         // 80 KB
  __shared__ v16bf sPackA[2 * 20 * 32];          // 40 KB (max: LSTM A 32x640)
  __shared__ unsigned long long sArg[BB];
  __shared__ int sTime[BB], sSym[BB], sLabel[BB], sEmit[BB], sLen[BB];

  int tid = threadIdx.x;
  int lane = tid & 31, wave = tid >> 5;
  __bf16* packS = (__bf16*)sPackA;

  // init: warm-up LSTM step from all-zero inputs (g = b_lstm)
  for (int e = tid; e < BB * HD; e += 512) {
    int jj = e % HD;
    float gi = sigm(b_lstm[jj]);
    float gg = tanhf(b_lstm[2 * HD + jj]);
    float go = sigm(b_lstm[3 * HD + jj]);
    float c0 = gi * gg;                 // f-gate * 0 + i * tanh(g)
    sC[e] = c0;
    sH[e] = go * tanhf(c0);
  }
  if (tid < BB) { sTime[tid] = 0; sSym[tid] = 0; sLen[tid] = 0; }
  __syncthreads();

  pred_update(sH, sDec, sPackA, pWpred, b_pred, tid, lane, wave);  // dec0

  for (int s = 0; s < SS; ++s) {
    // ---- phase 1: pack logits A = relu(enc_t + dec), 32x320 (Kt=10) ----
    for (int e = tid; e < 2 * 10 * 512; e += 512) {
      int i = e & 15, ln = (e >> 4) & 31, tile = e >> 9;
      int kt = tile % 10, mt = tile / 10;
      int m = mt * 16 + (ln & 15);
      int k = kt * 32 + kidxA(ln >> 4, i);
      int t0 = sTime[m]; if (t0 > TT - 1) t0 = TT - 1;   // gather clamp
      float v = xproj[((long)m * TT + t0) * JD + k] + sDec[m * JD + k];
      packS[e] = (__bf16)(v > 0.f ? v : 0.f);
    }
    if (tid < BB) sArg[tid] = 0ull;
    __syncthreads();

    // ---- phase 2: logits GEMM (32 x 1040) with fused row argmax ----
    for (int t = wave; t < 2 * 65; t += 16) {
      int mt = t & 1, nt = t >> 1;
      v8f acc = {};
      for (int kt = 0; kt < 10; ++kt) {
        v16bf a = sPackA[(mt * 10 + kt) * 32 + lane];
        const v16bf* bp = &((const v16bf*)pWout)[(nt * 10 + kt) * 32 + lane];
        if (kt + 1 < 10)
          __builtin_prefetch(&((const v16bf*)pWout)[(nt * 10 + kt + 1) * 32 + lane], 0, 1);
        acc = wmma_bf16(a, *bp, acc);
      }
      int n = nt * 16 + (lane & 15);
      bool valid = (n < NOUT);
      float bias = valid ? b_out[n] : 0.f;
      int mbase = mt * 16 + ((lane >> 4) << 3);
      for (int r = 0; r < 8; ++r) {
        float v = valid ? (acc[r] + bias) : -3.4e38f;
        int idx = n;
        for (int msk = 1; msk <= 8; msk <<= 1) {       // reduce over 16-lane N group
          float ov = __shfl_xor(v, msk, 32);
          int   oi = __shfl_xor(idx, msk, 32);
          if (ov > v || (ov == v && oi < idx)) { v = ov; idx = oi; }
        }
        if ((lane & 15) == 0) {
          unsigned int fb = __float_as_uint(v);
          unsigned int key = (fb & 0x80000000u) ? ~fb : (fb | 0x80000000u);
          unsigned long long pk =
              ((unsigned long long)key << 32) | (unsigned long long)(0xFFFFu - (unsigned)idx);
          atomicMax(&sArg[mbase + r], pk);
        }
      }
    }
    __syncthreads();

    // ---- phase 3: per-batch decision / outputs / time-sym update ----
    if (tid < BB) {
      int b = tid;
      unsigned long long pk = sArg[b];
      unsigned int key = (unsigned int)(pk >> 32);
      unsigned int fb = (key & 0x80000000u) ? (key & 0x7FFFFFFFu) : ~key;
      float score = __uint_as_float(fb);
      int label = 0xFFFF - (int)(pk & 0xFFFFu);
      int tcur = sTime[b];
      bool active = tcur < out_len[b];
      bool emit = active && (label != VV);
      sEmit[b] = emit ? 1 : 0;
      sLabel[b] = emit ? label : 0;                    // embedding gather index
      lab_out[s * BB + b] = emit ? label : VV;
      sc_out[s * BB + b] = active ? score : 0.f;
      tm_out[s * BB + b] = tcur;
      if (emit) sLen[b]++;
      int sym2 = emit ? sSym[b] + 1 : 0;
      bool force = emit && (sym2 >= 2);
      bool advance = (active && !emit) || force;
      sTime[b] = tcur + (advance ? 1 : 0);
      sSym[b] = advance ? 0 : sym2;
    }
    __syncthreads();

    // ---- phase 4: pack LSTM A = [emb | h], 32x640 (Kt=20) ----
    for (int e = tid; e < 2 * 20 * 512; e += 512) {
      int i = e & 15, ln = (e >> 4) & 31, tile = e >> 9;
      int kt = tile % 20, mt = tile / 20;
      int m = mt * 16 + (ln & 15);
      int k = kt * 32 + kidxA(ln >> 4, i);
      float v = (k < HD) ? Emb[(long)sLabel[m] * HD + k] : sH[m * HD + (k - HD)];
      packS[e] = (__bf16)v;
    }
    __syncthreads();

    // ---- phase 5: LSTM GEMM -> gates (32 x 1280), bias fused ----
    for (int t = wave; t < 2 * 80; t += 16) {
      int mt = t & 1, nt = t >> 1;
      v8f acc = {};
      for (int kt = 0; kt < 20; ++kt) {
        v16bf a = sPackA[(mt * 20 + kt) * 32 + lane];
        v16bf b = ((const v16bf*)pWlstm)[(nt * 20 + kt) * 32 + lane];
        acc = wmma_bf16(a, b, acc);
      }
      int n = nt * 16 + (lane & 15);
      float bias = b_lstm[n];
      int mbase = mt * 16 + ((lane >> 4) << 3);
      for (int r = 0; r < 8; ++r)
        sGates[(mbase + r) * (4 * HD) + n] = (__bf16)(acc[r] + bias);
    }
    __syncthreads();

    // ---- phase 6: elementwise LSTM state update (emit-masked) ----
    for (int e = tid; e < BB * HD; e += 512) {
      int b = e / HD, jj = e % HD;
      if (sEmit[b]) {
        const __bf16* g = &sGates[b * 4 * HD];
        float gi = sigm((float)g[jj]);
        float gf = sigm((float)g[HD + jj]);
        float gg = tanhf((float)g[2 * HD + jj]);
        float go = sigm((float)g[3 * HD + jj]);
        float c2 = gf * sC[e] + gi * gg;
        sC[e] = c2;
        sH[e] = go * tanhf(c2);
      }
    }
    __syncthreads();

    // ---- phase 7: dec = h @ W_pred + b_pred (invariant holds for non-emit rows) ----
    pred_update(sH, sDec, sPackA, pWpred, b_pred, tid, lane, wave);
  }

  if (tid < BB) len_out[tid] = sLen[tid];
}

// ---------------- host glue ----------------
extern "C" void kernel_launch(void* const* d_in, const int* in_sizes, int n_in,
                              void* d_out, int out_size, void* d_ws, size_t ws_size,
                              hipStream_t stream) {
  (void)in_sizes; (void)n_in; (void)out_size; (void)ws_size;
  const float* x      = (const float*)d_in[0];
  const int*   olen   = (const int*)d_in[1];
  const float* W_enc  = (const float*)d_in[2];
  const float* b_enc  = (const float*)d_in[3];
  const float* Emb    = (const float*)d_in[4];
  const float* W_ih   = (const float*)d_in[5];
  const float* W_hh   = (const float*)d_in[6];
  const float* b_lstm = (const float*)d_in[7];
  const float* W_pred = (const float*)d_in[8];
  const float* b_pred = (const float*)d_in[9];
  const float* W_out  = (const float*)d_in[10];
  const float* b_out  = (const float*)d_in[11];

  char* ws = (char*)d_ws;
  size_t off = 0;
  auto carve = [&](size_t bytes) {
    void* p = ws + off;
    off += (bytes + 255) & ~(size_t)255;
    return p;
  };
  float*  xproj     = (float*)carve(2048L * JD * 4);              // 2.62 MB
  __bf16* packX     = (__bf16*)carve(128L * 32 * 512 * 2);        // 4.19 MB
  __bf16* packWenc  = (__bf16*)carve(20L * 32 * 512 * 2);
  __bf16* packWout  = (__bf16*)carve(65L * 10 * 512 * 2);
  float*  lstmStage = (float*)carve(640L * 1280 * 4);             // 3.28 MB
  __bf16* packWlstm = (__bf16*)carve(80L * 20 * 512 * 2);
  __bf16* packWpred = (__bf16*)carve(20L * 10 * 512 * 2);

  // ---- one-time weight prep (cheap; re-run every call for determinism) ----
  concat_lstm_kernel<<<1024, 256, 0, stream>>>(W_ih, W_hh, lstmStage);
  pack_b_kernel<<<1024, 256, 0, stream>>>(W_enc,     packWenc,  ENC, JD,   JD);
  pack_b_kernel<<<1024, 256, 0, stream>>>(W_out,     packWout,  JD,  NOUT, NOUTP);
  pack_b_kernel<<<2048, 256, 0, stream>>>(lstmStage, packWlstm, 640, 1280, 1280);
  pack_b_kernel<<<1024, 256, 0, stream>>>(W_pred,    packWpred, HD,  JD,   JD);
  pack_a_kernel<<<2048, 256, 0, stream>>>(x, packX, 2048, ENC);

  // ---- encoder projection GEMM (WMMA bf16, f32 accumulate) ----
  enc_gemm_kernel<<<320, 256, 0, stream>>>(packX, packWenc, b_enc, xproj);

  // ---- persistent sequential greedy decode ----
  int*   lab = (int*)d_out;                 // (S, B) int32 bits
  float* sc  = (float*)d_out + SS * BB;     // (S, B) f32
  int*   tm  = (int*)d_out + 2 * SS * BB;   // (S, B) int32 bits
  int*   len = (int*)d_out + 3 * SS * BB;   // (B,)  int32 bits
  decode_kernel<<<1, 512, 0, stream>>>(xproj, olen, Emb, b_lstm, b_pred, b_out,
                                       packWout, packWlstm, packWpred,
                                       lab, sc, tm, len);
}
